// ResidualBlock_50173807952479
// MI455X (gfx1250) — compile-verified, matrix operand order checked by reading, not run on hardware
//
#include <hip/hip_runtime.h>
#include <hip/hip_bf16.h>
#include <stdint.h>

// ---------------------------------------------------------------------------
// Mamba residual block for MI455X (gfx1250, wave32, WMMA 16x16x32 f16).
// GEMMs (in_proj / x_proj / dt_proj / out_proj) run on v_wmma_f32_16x16x32_f16
// with f32 accumulation; conv/scan/gating are elementwise or register-recurrent.
// ---------------------------------------------------------------------------

#define B_SZ      4
#define L_SZ      2048
#define D_MODEL   1024
#define D_INNER   2048
#define D_STATE   16
#define D_CONV    4
#define DT_RANK   64
#define BL        (B_SZ * L_SZ)          // 8192 rows
#define EPS_RMS   1e-5f

typedef __attribute__((ext_vector_type(16))) _Float16 v16h;
typedef __attribute__((ext_vector_type(8)))  float    v8f;
typedef __attribute__((ext_vector_type(4)))  uint32_t v4u;

// ---------------------------------------------------------------------------
// WMMA fragment load (ISA 7.12.2 16-bit 16x32 layout). Per lane the fragment
// is two contiguous 16-byte chunks: halfs [hi*8, hi*8+7] and [16+hi*8, +7]
// => exactly two global_load_b128 from a hoisted per-lane base pointer.
// ---------------------------------------------------------------------------
__device__ inline v16h frag_from(const char* base, int koff_bytes) {
  union { v16h v; v4u q[2]; } f;
  f.q[0] = *reinterpret_cast<const v4u*>(base + koff_bytes);
  f.q[1] = *reinterpret_cast<const v4u*>(base + koff_bytes + 32);
  return f.v;
}

struct FragSet { v16h a0, a1, b0, b1, b2, b3; };

__device__ inline void load_set(FragSet& f, const char* a0p, const char* a1p,
                                const char* const* bp, int kb) {
  f.a0 = frag_from(a0p,   kb);
  f.a1 = frag_from(a1p,   kb);
  f.b0 = frag_from(bp[0], kb);
  f.b1 = frag_from(bp[1], kb);
  f.b2 = frag_from(bp[2], kb);
  f.b3 = frag_from(bp[3], kb);
}

__device__ inline void mma_set(v8f acc[2][4], const FragSet& f) {
  acc[0][0] = __builtin_amdgcn_wmma_f32_16x16x32_f16(false, f.a0, false, f.b0, (short)0, acc[0][0], false, false);
  acc[1][0] = __builtin_amdgcn_wmma_f32_16x16x32_f16(false, f.a1, false, f.b0, (short)0, acc[1][0], false, false);
  acc[0][1] = __builtin_amdgcn_wmma_f32_16x16x32_f16(false, f.a0, false, f.b1, (short)0, acc[0][1], false, false);
  acc[1][1] = __builtin_amdgcn_wmma_f32_16x16x32_f16(false, f.a1, false, f.b1, (short)0, acc[1][1], false, false);
  acc[0][2] = __builtin_amdgcn_wmma_f32_16x16x32_f16(false, f.a0, false, f.b2, (short)0, acc[0][2], false, false);
  acc[1][2] = __builtin_amdgcn_wmma_f32_16x16x32_f16(false, f.a1, false, f.b2, (short)0, acc[1][2], false, false);
  acc[0][3] = __builtin_amdgcn_wmma_f32_16x16x32_f16(false, f.a0, false, f.b3, (short)0, acc[0][3], false, false);
  acc[1][3] = __builtin_amdgcn_wmma_f32_16x16x32_f16(false, f.a1, false, f.b3, (short)0, acc[1][3], false, false);
}

// Epilogues: 0 = plain store, 1 = +bias then softplus, 2 = +residual
template <int EPI>
__device__ inline void store_tile(float* C, int ldc, int row0, int n, int hi,
                                  const v8f& c, int N, const float* bias,
                                  const float* res, int ldres) {
  if (n >= N) return;
#pragma unroll
  for (int r = 0; r < 8; ++r) {
    const int m = row0 + r + hi * 8;
    float v = c[r];
    if (EPI == 1) {                       // dt_proj: softplus(acc + bias)
      v += bias[n];
      v = (v > 20.f) ? v : log1pf(__expf(v));
    }
    if (EPI == 2) {                       // out_proj: + residual x
      v += res[(size_t)m * ldres + n];
    }
    C[(size_t)m * ldc + n] = v;
  }
}

// C[M,N] = A[M,K] * W[N,K]^T ; block tile 128x128, 8 waves (4 along M x 2
// along N), wave tile 32x64 = 2x4 accumulators, K step 32, unroll-by-2 with
// ping-pong fragment sets (no rotation copies; loads overlap the XDL pipe).
// Requires M % 128 == 0 and (K*2) % 128 == 0 (true at all call sites).
template <int EPI>
__global__ __launch_bounds__(256) void gemm_f16_wmma(
    const _Float16* __restrict__ A, const _Float16* __restrict__ W,
    float* __restrict__ C, int M, int N, int K, int ldc,
    const float* __restrict__ bias, const float* __restrict__ res, int ldres) {
  const int lane = threadIdx.x & 31;
  const int lo16 = lane & 15;
  const int hi   = lane >> 4;
  const int wave = threadIdx.x >> 5;
  const int wm = wave & 3;                      // 4 waves along M
  const int wn = wave >> 2;                     // 2 waves along N
  const int row0 = blockIdx.x * 128 + wm * 32;
  const int col0 = blockIdx.y * 128 + wn * 64;

  // Hoisted per-lane base pointers (bytes).
  const char* a0p = (const char*)(A + (size_t)(row0 + lo16)      * K) + hi * 16;
  const char* a1p = (const char*)(A + (size_t)(row0 + 16 + lo16) * K) + hi * 16;
  const char* bp[4];
  int ncol[4];
#pragma unroll
  for (int t = 0; t < 4; ++t) {
    int n = col0 + t * 16 + lo16;
    ncol[t] = n;
    n = (n < N) ? n : (N - 1);                  // clamp: branch-free K loop
    bp[t] = (const char*)(W + (size_t)n * K) + hi * 16;
  }

  v8f acc[2][4];
#pragma unroll
  for (int mi = 0; mi < 2; ++mi)
#pragma unroll
    for (int t = 0; t < 4; ++t) acc[mi][t] = v8f{};

  const int kend = K * 2;                       // bytes; multiple of 128

  FragSet f0, f1;
  load_set(f0, a0p, a1p, bp, 0);
  for (int kb = 0; kb < kend; kb += 128) {
    load_set(f1, a0p, a1p, bp, kb + 64);        // always in range (even steps)
    mma_set(acc, f0);
    if (kb + 128 < kend)                        // uniform; skips only last iter
      load_set(f0, a0p, a1p, bp, kb + 128);
    mma_set(acc, f1);
  }

#pragma unroll
  for (int mi = 0; mi < 2; ++mi)
#pragma unroll
    for (int t = 0; t < 4; ++t)
      store_tile<EPI>(C, ldc, row0 + mi * 16, ncol[t], hi, acc[mi][t], N, bias, res, ldres);
}

// ---------------------------------------------------------------------------
// RMSNorm (one 256-thread block per row of 1024) -> f16 activations.
// ---------------------------------------------------------------------------
__global__ __launch_bounds__(256) void rmsnorm_f16(const float* __restrict__ x,
                                                   const float* __restrict__ w,
                                                   _Float16* __restrict__ out) {
  __shared__ float sm[256];
  const int row = blockIdx.x;
  const float* xr = x + (size_t)row * D_MODEL;
  float ss = 0.f;
#pragma unroll
  for (int j = 0; j < D_MODEL / 256; ++j) {
    const float v = xr[threadIdx.x + j * 256];
    ss += v * v;
  }
  sm[threadIdx.x] = ss;
  __syncthreads();
  for (int s = 128; s > 0; s >>= 1) {
    if (threadIdx.x < s) sm[threadIdx.x] += sm[threadIdx.x + s];
    __syncthreads();
  }
  const float rs = rsqrtf(sm[0] / (float)D_MODEL + EPS_RMS);
  _Float16* orow = out + (size_t)row * D_MODEL;
#pragma unroll
  for (int j = 0; j < D_MODEL / 256; ++j) {
    const int i = threadIdx.x + j * 256;
    orow[i] = (_Float16)(xr[i] * rs * w[i]);
  }
}

__global__ void f32_to_f16(const float* __restrict__ src, _Float16* __restrict__ dst, int n) {
  const int i = blockIdx.x * blockDim.x + threadIdx.x;
  if (i < n) dst[i] = (_Float16)src[i];
}

// ---------------------------------------------------------------------------
// Depthwise causal conv (width 4) + SiLU. One thread per (b, channel),
// rolling register window over L. Reads xi = xz[:, 0:2048] (ld 4096).
// ---------------------------------------------------------------------------
__global__ __launch_bounds__(256) void conv_silu(const float* __restrict__ xz,
                                                 const float* __restrict__ cw,
                                                 const float* __restrict__ cb,
                                                 float* __restrict__ u,
                                                 _Float16* __restrict__ uf16) {
  const int idx = blockIdx.x * blockDim.x + threadIdx.x;
  if (idx >= B_SZ * D_INNER) return;
  const int b = idx / D_INNER;
  const int d = idx % D_INNER;
  const float w0 = cw[d * D_CONV + 0], w1 = cw[d * D_CONV + 1];
  const float w2 = cw[d * D_CONV + 2], w3 = cw[d * D_CONV + 3];
  const float bias = cb[d];
  float x0 = 0.f, x1 = 0.f, x2 = 0.f;
  for (int l = 0; l < L_SZ; ++l) {
    const size_t r = (size_t)(b * L_SZ + l);
    const float x3 = xz[r * 4096 + d];
    float v = w0 * x0 + w1 * x1 + w2 * x2 + w3 * x3 + bias;
    v = v / (1.f + __expf(-v));                             // SiLU
    u[r * D_INNER + d]    = v;
    uf16[r * D_INNER + d] = (_Float16)v;
    x0 = x1; x1 = x2; x2 = x3;
  }
}

// dt columns of x_dbl -> f16 for the dt_proj GEMM.
__global__ void dt_extract(const float* __restrict__ xdbl, _Float16* __restrict__ dtf16) {
  const int i = blockIdx.x * blockDim.x + threadIdx.x;
  if (i >= BL * DT_RANK) return;
  const int r = i / DT_RANK, j = i % DT_RANK;
  dtf16[i] = (_Float16)xdbl[(size_t)r * 96 + j];
}

// ---------------------------------------------------------------------------
// Selective scan: thread = one (b, d) channel, 16-state register recurrence.
// B/C (32 floats per step, shared across all d) staged via LDS in chunks.
// delta lives in xz cols [0,2048) with ld 4096; y overwrites u in place.
// ---------------------------------------------------------------------------
__global__ __launch_bounds__(256) void selective_scan(
    const float* __restrict__ delta,        // xz base, ld 4096
    const float* __restrict__ xdbl,         // [BL, 96]; B at col 64, C at col 80
    const float* __restrict__ A_log,        // [D_INNER, 16]
    const float* __restrict__ Dp,           // [D_INNER]
    float* __restrict__ u) {                // in: u ; out: y (in place)
  const int b = blockIdx.x >> 3;                       // 8 blocks per batch
  const int d = ((blockIdx.x & 7) << 8) + threadIdx.x; // 256 channels / block
  float a[D_STATE], s[D_STATE];
#pragma unroll
  for (int i = 0; i < D_STATE; ++i) {
    a[i] = -__expf(A_log[(size_t)d * D_STATE + i]);
    s[i] = 0.f;
  }
  const float Dd = Dp[d];
  __shared__ float bc[64][32];                         // 64 steps of B|C
  for (int l0 = 0; l0 < L_SZ; l0 += 64) {
    for (int t = threadIdx.x; t < 64 * 32; t += 256) {
      const int li = t >> 5, j = t & 31;
      bc[li][j] = xdbl[((size_t)(b * L_SZ + l0 + li)) * 96 + 64 + j];
    }
    __syncthreads();
    for (int li = 0; li < 64; ++li) {
      const size_t r = (size_t)(b * L_SZ + l0 + li);
      const float dt = delta[r * 4096 + d];
      const float ut = u[r * D_INNER + d];
      float y = 0.f;
#pragma unroll
      for (int i = 0; i < D_STATE; ++i) {
        const float dA = __expf(dt * a[i]);
        s[i] = s[i] * dA + dt * bc[li][i] * ut;
        y += s[i] * bc[li][D_STATE + i];
      }
      u[r * D_INNER + d] = y + ut * Dd;
    }
    __syncthreads();
  }
}

// y * silu(z) -> f16 for out_proj. z = xz cols [2048, 4096).
__global__ void gate_silu(const float* __restrict__ y, const float* __restrict__ xz,
                          _Float16* __restrict__ yf16) {
  const int i = blockIdx.x * blockDim.x + threadIdx.x;
  if (i >= BL * D_INNER) return;
  const int r = i / D_INNER, d = i % D_INNER;
  const float z = xz[(size_t)r * 4096 + D_INNER + d];
  yf16[i] = (_Float16)(y[i] * z / (1.f + __expf(-z)));
}

// ---------------------------------------------------------------------------
// Launch: workspace layout (bytes, all 256-aligned):
//   hf16   [BL*1024 f16]  @ 0         (16 MB)
//   wbuf   [<=4096*1024]  @ 16 MB     ( 8 MB, reused per weight)
//   xz     [BL*4096 f32]  @ 24 MB     (128 MB; xi cols reused for delta)
//   u/y    [BL*2048 f32]  @ 152 MB    (64 MB; y written in place)
//   xif16  [BL*2048 f16]  @ 216 MB    (32 MB; reused for gated-y f16)
//   xdbl   [BL*96 f32]    @ 248 MB    ( 3 MB)
//   dtf16  [BL*64 f16]    @ 251 MB    ( 1 MB)      total ~252 MB
// ---------------------------------------------------------------------------
extern "C" void kernel_launch(void* const* d_in, const int* in_sizes, int n_in,
                              void* d_out, int out_size, void* d_ws, size_t ws_size,
                              hipStream_t stream) {
  const float* x         = (const float*)d_in[0];
  const float* norm_w    = (const float*)d_in[1];
  const float* in_proj_w = (const float*)d_in[2];
  const float* conv_w    = (const float*)d_in[3];
  const float* conv_b    = (const float*)d_in[4];
  const float* x_proj_w  = (const float*)d_in[5];
  const float* dt_proj_w = (const float*)d_in[6];
  const float* dt_proj_b = (const float*)d_in[7];
  const float* A_log     = (const float*)d_in[8];
  const float* Dp        = (const float*)d_in[9];
  const float* out_proj_w= (const float*)d_in[10];
  float* out = (float*)d_out;

  char* ws = (char*)d_ws;
  _Float16* hf16  = (_Float16*)(ws);
  _Float16* wbuf  = (_Float16*)(ws + (16u << 20));
  float*    xz    = (float*)   (ws + (24u << 20));
  float*    u     = (float*)   (ws + (152u << 20));
  _Float16* xif16 = (_Float16*)(ws + (216u << 20));
  float*    xdbl  = (float*)   (ws + (248u << 20));
  _Float16* dtf16 = (_Float16*)(ws + (251u << 20));

  // 1. RMSNorm -> f16 activations
  rmsnorm_f16<<<BL, 256, 0, stream>>>(x, norm_w, hf16);

  // 2. in_proj: xz[BL,4096] = h * in_proj_w^T
  {
    int n = 2 * D_INNER * D_MODEL;
    f32_to_f16<<<(n + 255) / 256, 256, 0, stream>>>(in_proj_w, wbuf, n);
    dim3 g(BL / 128, (2 * D_INNER + 127) / 128);
    gemm_f16_wmma<0><<<g, 256, 0, stream>>>(hf16, wbuf, xz, BL, 2 * D_INNER, D_MODEL,
                                            2 * D_INNER, nullptr, nullptr, 0);
  }

  // 3. depthwise conv + SiLU -> u (f32) and xif16
  conv_silu<<<(B_SZ * D_INNER) / 256, 256, 0, stream>>>(xz, conv_w, conv_b, u, xif16);

  // 4. x_proj: xdbl[BL,96] = xi * x_proj_w^T
  {
    int n = (DT_RANK + 2 * D_STATE) * D_INNER;
    f32_to_f16<<<(n + 255) / 256, 256, 0, stream>>>(x_proj_w, wbuf, n);
    dim3 g(BL / 128, 1);
    gemm_f16_wmma<0><<<g, 256, 0, stream>>>(xif16, wbuf, xdbl, BL, 96, D_INNER,
                                            96, nullptr, nullptr, 0);
  }

  // 5. dt_proj + softplus: delta (stored into xz cols [0,2048), ld 4096)
  dt_extract<<<(BL * DT_RANK + 255) / 256, 256, 0, stream>>>(xdbl, dtf16);
  {
    int n = D_INNER * DT_RANK;
    f32_to_f16<<<(n + 255) / 256, 256, 0, stream>>>(dt_proj_w, wbuf, n);
    dim3 g(BL / 128, (D_INNER + 127) / 128);
    gemm_f16_wmma<1><<<g, 256, 0, stream>>>(dtf16, wbuf, xz, BL, D_INNER, DT_RANK,
                                            2 * D_INNER, dt_proj_b, nullptr, 0);
  }

  // 6. selective scan (y overwrites u)
  selective_scan<<<B_SZ * (D_INNER / 256), 256, 0, stream>>>(xz, xdbl, A_log, Dp, u);

  // 7. gate: yf16 = f16(y * silu(z))  (reuses xif16 slab)
  gate_silu<<<(BL * D_INNER + 255) / 256, 256, 0, stream>>>(u, xz, xif16);

  // 8. out_proj + residual: out = y_gated * out_proj_w^T + x
  {
    int n = D_MODEL * D_INNER;
    f32_to_f16<<<(n + 255) / 256, 256, 0, stream>>>(out_proj_w, wbuf, n);
    dim3 g(BL / 128, (D_MODEL + 127) / 128);
    gemm_f16_wmma<2><<<g, 256, 0, stream>>>(xif16, wbuf, out, BL, D_MODEL, D_INNER,
                                            D_MODEL, nullptr, x, D_MODEL);
  }
}
